// m2oAttentionA2_32177894982192
// MI455X (gfx1250) — matrix operand order repaired and checked by reading, not compiled, not run on hardware
//
#include <hip/hip_runtime.h>
#include <hip/hip_bf16.h>

// ---- problem constants (match reference) ----
#define BDIM   1024
#define SDIM   200
#define SPAD   208      // 13 tiles of 16
#define KDIM   128
#define HDIM   128
#define LDA    136      // bf16 elements per sA row (128 + 8 pad, 272B = odd*16B)
#define LDB    136      // bf16 elements per sW1t row
#define NTHREADS 256

typedef __attribute__((ext_vector_type(16))) __bf16 v16bf;
typedef __attribute__((ext_vector_type(8)))  __bf16 bf16x8;
typedef __attribute__((ext_vector_type(4)))  __bf16 bf16x4;
typedef __attribute__((ext_vector_type(8)))  float  v8f;

__global__ __launch_bounds__(NTHREADS)
void m2o_attn_fused_kernel(const float* __restrict__ input,   // [B,S,128]
                           const float* __restrict__ z,       // [B,128]
                           const int*   __restrict__ mask,    // [B,S]
                           const float* __restrict__ W1,      // [128,128]
                           const float* __restrict__ W2,      // [128,128]
                           const float* __restrict__ b2,      // [128]
                           const float* __restrict__ V,       // [128,1]
                           const float* __restrict__ bV,      // [1]
                           float* __restrict__ out)           // [B,128]
{
    extern __shared__ __align__(16) char smem[];
    __bf16* sA    = (__bf16*)smem;                 // SPAD * LDA bf16
    __bf16* sW1t  = sA + SPAD * LDA;               // 128 * LDB bf16 (transposed W1: [n][k])
    float*  sW2h  = (float*)(sW1t + HDIM * LDB);   // 128
    float*  sV    = sW2h + HDIM;                   // 128
    float*  sScore= sV + HDIM;                     // SPAD
    float*  sPool = sScore + SPAD;                 // 128
    float*  sRed  = sPool + HDIM;                  // 16

    const int t    = threadIdx.x;
    const int b    = blockIdx.x;
    const int lane = t & 31;
    const int wid  = t >> 5;
    const int half = lane >> 4;        // 0: lanes 0-15, 1: lanes 16-31
    const int ln   = lane & 15;

    const float bVv = bV[0];

    // ---------------- Stage 1: load input[b] (f32) -> sA (bf16), once from HBM ----
    {
        const float4* src = (const float4*)(input + (size_t)b * SDIM * KDIM);
        // 200*128/4 = 6400 float4
        for (int i = t; i < (SDIM * KDIM / 4); i += NTHREADS) {
            float4 v = src[i];
            int s  = i >> 5;            // 32 float4 per row
            int c4 = (i & 31) << 2;
            bf16x4 p;
            p[0] = (__bf16)v.x; p[1] = (__bf16)v.y;
            p[2] = (__bf16)v.z; p[3] = (__bf16)v.w;
            *(bf16x4*)(sA + s * LDA + c4) = p;      // ds_store_b64
        }
        // zero padding rows 200..207 (contiguous region)
        for (int i = t; i < (SPAD - SDIM) * LDA; i += NTHREADS)
            sA[SDIM * LDA + i] = (__bf16)0.0f;
    }

    // ---------------- Stage 2: W1 (f32, [k][n]) -> sW1t (bf16, [n][k]) ------------
    for (int i = t; i < (KDIM * HDIM / 4); i += NTHREADS) {
        float4 v = ((const float4*)W1)[i];
        int k = i >> 5;
        int n = (i & 31) << 2;
        sW1t[(n + 0) * LDB + k] = (__bf16)v.x;
        sW1t[(n + 1) * LDB + k] = (__bf16)v.y;
        sW1t[(n + 2) * LDB + k] = (__bf16)v.z;
        sW1t[(n + 3) * LDB + k] = (__bf16)v.w;
    }

    // ---------------- Stage 3: w2h = z[b] @ W2 + b2 ; copy V ----------------------
    if (t < HDIM) {
        float acc = b2[t];
        const float* zb = z + (size_t)b * KDIM;
        for (int d = 0; d < KDIM; ++d)
            acc = fmaf(zb[d], W2[d * HDIM + t], acc);
        sW2h[t] = acc;
    } else {
        int h = t - HDIM;               // 0..127
        sV[h] = V[h];
    }

    __syncthreads();

    // ---------------- Stage 4: WMMA GEMM + tanh + V-dot -> raw scores -------------
    // wave 'wid' handles m-tiles wid, wid+8 (13 tiles total)
    for (int mt = wid; mt < SPAD / 16; mt += 8) {
        float rowacc[8];
        #pragma unroll
        for (int r = 0; r < 8; ++r) rowacc[r] = 0.0f;

        const int arow = (mt * 16 + ln) * LDA;

        for (int nt = 0; nt < HDIM / 16; ++nt) {
            v8f c = {};
            const int nn = nt * 16 + ln;

            #pragma unroll
            for (int kk = 0; kk < KDIM / 32; ++kk) {
                const int k0 = kk * 32;
                // A fragment: lanes 0-15 K{0-7,16-23}, lanes 16-31 K{8-15,24-31}
                bf16x8 a0 = *(const bf16x8*)(sA + arow + k0 + half * 8);
                bf16x8 a1 = *(const bf16x8*)(sA + arow + k0 + 16 + half * 8);
                // B fragment: lanes 0-15 K{0-7,16-23}, lanes 16-31 K{8-15,24-31}
                const __bf16* bp = sW1t + nn * LDB + k0 + half * 8;
                bf16x8 b0 = *(const bf16x8*)(bp);
                bf16x8 b1 = *(const bf16x8*)(bp + 16);

                v16bf a, bb;
                #pragma unroll
                for (int i = 0; i < 8; ++i) {
                    a[i] = a0[i];  a[i + 8] = a1[i];
                    bb[i] = b0[i]; bb[i + 8] = b1[i];
                }
                c = __builtin_amdgcn_wmma_f32_16x16x32_bf16(
                        false, a, false, bb, (short)0, c, false, false);
            }

            const float wh  = sW2h[nn];
            const float vv  = sV[nn];
            const float vv2 = 2.0f * vv;
            #pragma unroll
            for (int r = 0; r < 8; ++r) {
                float x = c[r] + wh;
                // tanh(x) = 1 - 2/(e^{2x}+1); fast rcp (TRANS pipe, co-execs w/ WMMA)
                float e = __expf(2.0f * x);
                float rcp = __builtin_amdgcn_rcpf(e + 1.0f);
                rowacc[r] = fmaf(-vv2, rcp, rowacc[r] + vv);
            }
        }

        // reduce over N: sum across each 16-lane half (rows differ between halves)
        #pragma unroll
        for (int r = 0; r < 8; ++r) {
            float v = rowacc[r];
            v += __shfl_xor(v, 1, 32);
            v += __shfl_xor(v, 2, 32);
            v += __shfl_xor(v, 4, 32);
            v += __shfl_xor(v, 8, 32);
            rowacc[r] = v;
        }
        if (ln == 0) {
            const int rb = mt * 16 + half * 8;   // lanes 0/16 own rows rb..rb+7
            #pragma unroll
            for (int r = 0; r < 8; ++r)
                sScore[rb + r] = rowacc[r] + bVv;
        }
    }
    __syncthreads();

    // ---------------- Stage 5: masked softmax over S ------------------------------
    float sc = -3.0e38f;
    if (t < SDIM) {
        sc = sScore[t];
        if (mask[(size_t)b * SDIM + t] == 0) sc = -1.0e10f;
    }
    // block max
    {
        float m = sc;
        m = fmaxf(m, __shfl_xor(m, 16, 32));
        m = fmaxf(m, __shfl_xor(m, 8, 32));
        m = fmaxf(m, __shfl_xor(m, 4, 32));
        m = fmaxf(m, __shfl_xor(m, 2, 32));
        m = fmaxf(m, __shfl_xor(m, 1, 32));
        if (lane == 0) sRed[wid] = m;
    }
    __syncthreads();
    if (t == 0) {
        float mm = sRed[0];
        for (int i = 1; i < 8; ++i) mm = fmaxf(mm, sRed[i]);
        sRed[8] = mm;
    }
    __syncthreads();
    const float bmax = sRed[8];
    float e = (t < SDIM) ? __expf(sc - bmax) : 0.0f;
    {
        float s = e;
        s += __shfl_xor(s, 16, 32);
        s += __shfl_xor(s, 8, 32);
        s += __shfl_xor(s, 4, 32);
        s += __shfl_xor(s, 2, 32);
        s += __shfl_xor(s, 1, 32);
        if (lane == 0) sRed[wid] = s;
    }
    __syncthreads();
    if (t == 0) {
        float ss = 0.0f;
        for (int i = 0; i < 8; ++i) ss += sRed[i];
        sRed[9] = ss;
    }
    __syncthreads();
    const float inv = __builtin_amdgcn_rcpf(sRed[9]);
    if (t < SDIM) sScore[t] = e * inv;
    __syncthreads();

    // ---------------- Stage 6: out[b,d] = sum_s att[s]*input[b,s,d] (from LDS) ----
    {
        const int d = t & 127;
        const int g = t >> 7;               // 0 or 1: split S range
        float acc = 0.0f;
        const int s0 = g * (SDIM / 2);
        for (int s = s0; s < s0 + SDIM / 2; ++s)
            acc = fmaf(sScore[s], (float)sA[s * LDA + d], acc);
        if (g == 1) sPool[d] = acc;
        __syncthreads();
        if (g == 0)
            out[(size_t)b * HDIM + d] = acc + sPool[d];
    }
}

extern "C" void kernel_launch(void* const* d_in, const int* in_sizes, int n_in,
                              void* d_out, int out_size, void* d_ws, size_t ws_size,
                              hipStream_t stream) {
    const float* input = (const float*)d_in[0];
    const float* z     = (const float*)d_in[1];
    const int*   mask  = (const int*)d_in[2];
    const float* W1    = (const float*)d_in[3];
    const float* W2    = (const float*)d_in[4];
    const float* b2    = (const float*)d_in[5];
    const float* V     = (const float*)d_in[6];
    const float* bV    = (const float*)d_in[7];
    float* out = (float*)d_out;

    const size_t smem =
        (size_t)(SPAD * LDA + HDIM * LDB) * sizeof(__bf16) +   // sA + sW1t
        (size_t)(HDIM + HDIM + SPAD + HDIM + 16) * sizeof(float);

    m2o_attn_fused_kernel<<<BDIM, NTHREADS, smem, stream>>>(
        input, z, mask, W1, W2, b2, V, bV, out);
}
